// HIGSyn_91122026152855
// MI455X (gfx1250) — matrix operand order repaired and checked by reading, not compiled, not run on hardware
//
#include <hip/hip_runtime.h>
#include <math.h>

// ---------------------------------------------------------------------------
// Problem constants (from the reference)
// ---------------------------------------------------------------------------
#define NN   50000      // nodes
#define EE   1600000    // edges
#define DIN  78         // atom feature dim
#define DPAD 96         // DIN padded to multiple of 32 for WMMA K
#define HD   128        // gnn dim
#define HD2  256        // hidden of GIN MLP
#define GG   1024       // graphs
#define BNE  1e-5f

typedef __bf16 bf16;
typedef __attribute__((ext_vector_type(16))) bf16  bf16x16;
typedef __attribute__((ext_vector_type(8)))  float f32x8;

union FragU { bf16x16 v; uint4 u[2]; };

// ---------------------------------------------------------------------------
// Elementwise / prep kernels
// ---------------------------------------------------------------------------
__global__ void k_zero_f(float* __restrict__ p, int n) {
  int i = blockIdx.x * blockDim.x + threadIdx.x;
  if (i < n) p[i] = 0.0f;
}

// eval BatchNorm1d: y = g*(x-m)*rsqrt(v+eps)+b ; bn rows: g,b,m,v
__global__ void k_bn_apply(const float* __restrict__ x, const float* __restrict__ bn,
                           float* __restrict__ y, int total, int C) {
  int i = blockIdx.x * blockDim.x + threadIdx.x;
  if (i >= total) return;
  int c = i % C;
  float g = bn[c], b = bn[C + c], m = bn[2 * C + c], v = bn[3 * C + c];
  y[i] = g * (x[i] - m) * rsqrtf(v + BNE) + b;
}

// transpose-convert weights: wt[c*Kpad + k] = bf16(w[k*C + c]), zero pad K
__global__ void k_prep_wT(const float* __restrict__ w, bf16* __restrict__ wt,
                          int K, int Kpad, int C) {
  int i = blockIdx.x * blockDim.x + threadIdx.x;
  if (i >= C * Kpad) return;
  int c = i / Kpad, k = i % Kpad;
  wt[i] = (k < K) ? (bf16)w[(size_t)k * C + c] : (bf16)0.0f;
}

// fold BN into affine: scale = g*rsqrt(v+eps), shift = b - m*scale
__global__ void k_prep_bn(const float* __restrict__ bn, float* __restrict__ scale,
                          float* __restrict__ shift, int C) {
  int c = blockIdx.x * blockDim.x + threadIdx.x;
  if (c >= C) return;
  float g = bn[c], b = bn[C + c], m = bn[2 * C + c], v = bn[3 * C + c];
  float s = g * rsqrtf(v + BNE);
  scale[c] = s;
  shift[c] = b - m * s;
}

// GIN aggregation: agg[dst] += feat[src]  (one thread per edge*channel)
__global__ void k_agg_edges(const float* __restrict__ feat, const int* __restrict__ src,
                            const int* __restrict__ dst, float* __restrict__ agg,
                            int total, int C) {
  int i = blockIdx.x * blockDim.x + threadIdx.x;
  if (i >= total) return;
  int e = i / C, c = i % C;
  atomicAdd(&agg[(size_t)dst[e] * C + c], feat[(size_t)src[e] * C + c]);
}

// A-operand build: abf = bf16(h + agg), K zero-padded to Cpad
__global__ void k_add_cvt(const float* __restrict__ h, const float* __restrict__ agg,
                          bf16* __restrict__ out, int total, int C, int Cpad) {
  int i = blockIdx.x * blockDim.x + threadIdx.x;
  if (i >= total) return;
  int n = i / Cpad, c = i % Cpad;
  out[i] = (c < C) ? (bf16)(h[(size_t)n * C + c] + agg[(size_t)n * C + c]) : (bf16)0.0f;
}

// ---------------------------------------------------------------------------
// WMMA GEMM:  D[M,Ncols] = A[M,K](bf16) * W[K,Ncols](bf16, stored as BT[Ncols,K])
// MODE 0: out = bf16(relu(acc + bias))            (first MLP GEMM)
// MODE 1: out = f32(relu(acc + bias)*scale+shift) (second MLP GEMM + BN fold)
// Wave: one 16x32 output slab, 2 accumulators sharing one A fragment.
// Block: 4 waves -> 128 columns. Requires M%16==0, K%32==0, Ncols%128==0.
// Operand layouts: 16-bit A 16x32, B 32x16 in contiguous k (k = 16*(lane/16) + s); f32 C/D 16x16 per ISA.
// ---------------------------------------------------------------------------
template <int MODE>
__global__ __launch_bounds__(128) void k_wmma_gemm(
    const bf16* __restrict__ A, int lda,
    const bf16* __restrict__ BT, int ldbt,
    const float* __restrict__ bias,
    const float* __restrict__ scale, const float* __restrict__ shift,
    bf16* __restrict__ outB, float* __restrict__ outF, int ldc,
    int M, int K, int Ncols) {
  const int lane = threadIdx.x & 31;
  const int wave = threadIdx.x >> 5;
  const int m0 = blockIdx.x * 16;
  const int n0 = (blockIdx.y * 4 + wave) * 32;
  if (m0 >= M || n0 >= Ncols) return;

  const int row = m0 + (lane & 15);
  const int col = n0 + (lane & 15);
  const int aK = (lane >> 4) << 4;   // lanes 16-31 hold K+16..31
  const int bK = (lane >> 4) << 4;   // lanes 16-31 hold K+16..31

  const bf16* ap  = A  + (size_t)row * lda;
  const bf16* bp0 = BT + (size_t)col * ldbt;
  const bf16* bp1 = BT + (size_t)(col + 16) * ldbt;

  f32x8 acc0 = {};
  f32x8 acc1 = {};
  for (int k0 = 0; k0 < K; k0 += 32) {
    FragU a, b0, b1;
    a.u[0]  = *reinterpret_cast<const uint4*>(ap + k0 + aK);        // K: +0..7  (or +16..23)
    a.u[1]  = *reinterpret_cast<const uint4*>(ap + k0 + aK + 8);    // K: +8..15 (or +24..31)
    b0.u[0] = *reinterpret_cast<const uint4*>(bp0 + k0 + bK);
    b0.u[1] = *reinterpret_cast<const uint4*>(bp0 + k0 + bK + 8);
    b1.u[0] = *reinterpret_cast<const uint4*>(bp1 + k0 + bK);
    b1.u[1] = *reinterpret_cast<const uint4*>(bp1 + k0 + bK + 8);
    acc0 = __builtin_amdgcn_wmma_f32_16x16x32_bf16(false, a.v, false, b0.v,
                                                   (short)0, acc0, false, false);
    acc1 = __builtin_amdgcn_wmma_f32_16x16x32_bf16(false, a.v, false, b1.v,
                                                   (short)0, acc1, false, false);
  }

  const int rbase = m0 + ((lane >> 4) << 3);
  const float bias0 = bias[col];
  const float bias1 = bias[col + 16];
#pragma unroll
  for (int r = 0; r < 8; ++r) {
    int rr = rbase + r;
    float y0 = acc0[r] + bias0;
    float y1 = acc1[r] + bias1;
    y0 = y0 > 0.0f ? y0 : 0.0f;
    y1 = y1 > 0.0f ? y1 : 0.0f;
    if (MODE == 0) {
      outB[(size_t)rr * ldc + col]      = (bf16)y0;
      outB[(size_t)rr * ldc + col + 16] = (bf16)y1;
    } else {
      outF[(size_t)rr * ldc + col]      = y0 * scale[col]      + shift[col];
      outF[(size_t)rr * ldc + col + 16] = y1 * scale[col + 16] + shift[col + 16];
    }
  }
}

// ---------------------------------------------------------------------------
// GCN-score SAG pooling pieces
// ---------------------------------------------------------------------------
__global__ void k_deg_init(float* __restrict__ d, int n) {
  int i = blockIdx.x * blockDim.x + threadIdx.x;
  if (i < n) d[i] = 1.0f;                       // self loop
}
__global__ void k_deg_edge(const int* __restrict__ dst, float* __restrict__ d, int E) {
  int i = blockIdx.x * blockDim.x + threadIdx.x;
  if (i < E) atomicAdd(&d[dst[i]], 1.0f);
}
__global__ void k_deg_fin(float* __restrict__ d, int n) {   // in-place -> dinv
  int i = blockIdx.x * blockDim.x + threadIdx.x;
  if (i < n) d[i] = rsqrtf(d[i]);
}

// xw[n] = sum_c h[n,c]*w[c]  (accum=1 adds into xw)
__global__ void k_node_dot(const float* __restrict__ h, const float* __restrict__ w,
                           float* __restrict__ xw, int N, int C, int accum) {
  int n = blockIdx.x * blockDim.x + threadIdx.x;
  if (n >= N) return;
  float s = 0.0f;
  const float* row = h + (size_t)n * C;
  for (int c = 0; c < C; ++c) s += row[c] * w[c];
  xw[n] = accum ? (xw[n] + s) : s;
}

__global__ void k_score_init(const float* __restrict__ xw, const float* __restrict__ dinv,
                             float* __restrict__ s, int N) {
  int n = blockIdx.x * blockDim.x + threadIdx.x;
  if (n < N) s[n] = xw[n] * dinv[n] * dinv[n];      // self-loop message
}
__global__ void k_score_edge(const float* __restrict__ xw, const float* __restrict__ dinv,
                             const int* __restrict__ src, const int* __restrict__ dst,
                             float* __restrict__ s, int E) {
  int e = blockIdx.x * blockDim.x + threadIdx.x;
  if (e >= E) return;
  int si = src[e], di = dst[e];
  atomicAdd(&s[di], xw[si] * dinv[si] * dinv[di]);
}
// s <- exp(tanh(s+b)); z[g] += s   (tanh in [-1,1] -> max-subtraction unneeded)
__global__ void k_score_fin(float* __restrict__ s, const float* __restrict__ bptr,
                            const int* __restrict__ batch, float* __restrict__ z, int N) {
  int n = blockIdx.x * blockDim.x + threadIdx.x;
  if (n >= N) return;
  float t = expf(tanhf(s[n] + bptr[0]));
  s[n] = t;
  atomicAdd(&z[batch[n]], t);
}
// out[g, off+c] += h[n,c] * e[n]/z[g]
__global__ void k_pool_add(const float* __restrict__ h, const float* __restrict__ e,
                           const float* __restrict__ z, const int* __restrict__ batch,
                           float* __restrict__ out, int total, int C, int ldo, int off) {
  int i = blockIdx.x * blockDim.x + threadIdx.x;
  if (i >= total) return;
  int n = i / C, c = i % C;
  int g = batch[n];
  atomicAdd(&out[(size_t)g * ldo + off + c], h[i] * (e[n] / z[g]));
}

// ---------------------------------------------------------------------------
// Host
// ---------------------------------------------------------------------------
static inline int cdiv(int a, int b) { return (a + b - 1) / b; }

extern "C" void kernel_launch(void* const* d_in, const int* in_sizes, int n_in,
                              void* d_out, int out_size, void* d_ws, size_t ws_size,
                              hipStream_t stream) {
  (void)in_sizes; (void)n_in;
  const float* x    = (const float*)d_in[0];
  const int*   ei   = (const int*)d_in[1];
  const int*   src  = ei;
  const int*   dst  = ei + EE;
  const int*   batch= (const int*)d_in[2];
  const float* bn0  = (const float*)d_in[3];
  const float* w1a  = (const float*)d_in[4];
  const float* b1a  = (const float*)d_in[5];
  const float* w1b  = (const float*)d_in[6];
  const float* b1b  = (const float*)d_in[7];
  const float* bn1  = (const float*)d_in[8];
  const float* w2a  = (const float*)d_in[9];
  const float* b2a  = (const float*)d_in[10];
  const float* w2b  = (const float*)d_in[11];
  const float* b2b  = (const float*)d_in[12];
  const float* bn2  = (const float*)d_in[13];
  const float* w3a  = (const float*)d_in[14];
  const float* b3a  = (const float*)d_in[15];
  const float* w3b  = (const float*)d_in[16];
  const float* b3b  = (const float*)d_in[17];
  const float* bn3  = (const float*)d_in[18];
  const float* wp1  = (const float*)d_in[19];
  const float* bp1  = (const float*)d_in[20];
  const float* wp2  = (const float*)d_in[21];
  const float* bp2  = (const float*)d_in[22];
  const float* wp3  = (const float*)d_in[23];
  const float* bp3  = (const float*)d_in[24];
  const float* wp4  = (const float*)d_in[25];
  const float* bp4  = (const float*)d_in[26];

  // ---- workspace carve-out (256B aligned) ----
  char* ws = (char*)d_ws;
  size_t off = 0;
  auto take = [&](size_t bytes) -> char* {
    char* p = ws + off;
    off = (off + bytes + 255) & ~(size_t)255;
    return p;
  };
  float* h1     = (float*)take((size_t)NN * HD * 4);
  float* h2     = (float*)take((size_t)NN * HD * 4);
  float* h3     = (float*)take((size_t)NN * HD * 4);
  float* xbn    = (float*)take((size_t)NN * DIN * 4);
  float* agg    = (float*)take((size_t)NN * HD * 4);
  bf16*  abf    = (bf16*) take((size_t)NN * HD * 2);
  bf16*  midbf  = (bf16*) take((size_t)NN * HD2 * 2);
  bf16*  w1aT   = (bf16*) take((size_t)HD2 * DPAD * 2);
  bf16*  w1bT   = (bf16*) take((size_t)HD  * HD2 * 2);
  bf16*  w2aT   = (bf16*) take((size_t)HD2 * HD  * 2);
  bf16*  w2bT   = (bf16*) take((size_t)HD  * HD2 * 2);
  bf16*  w3aT   = (bf16*) take((size_t)HD2 * HD  * 2);
  bf16*  w3bT   = (bf16*) take((size_t)HD  * HD2 * 2);
  float* scale1 = (float*)take(HD * 4);
  float* shift1 = (float*)take(HD * 4);
  float* scale2 = (float*)take(HD * 4);
  float* shift2 = (float*)take(HD * 4);
  float* scale3 = (float*)take(HD * 4);
  float* shift3 = (float*)take(HD * 4);
  float* dinv   = (float*)take((size_t)NN * 4);
  float* xw     = (float*)take((size_t)NN * 4);
  float* sbuf   = (float*)take((size_t)NN * 4);
  float* zbuf   = (float*)take((size_t)GG * 4);
  if (off > ws_size) return;  // insufficient scratch; bail deterministically

  const int TB = 256;
  float* out = (float*)d_out;

  // zero output (harness poisons it)
  k_zero_f<<<cdiv(out_size, TB), TB, 0, stream>>>(out, out_size);

  // BN0 on node features
  k_bn_apply<<<cdiv(NN * DIN, TB), TB, 0, stream>>>(x, bn0, xbn, NN * DIN, DIN);

  // weight / BN prep
  k_prep_wT<<<cdiv(HD2 * DPAD, TB), TB, 0, stream>>>(w1a, w1aT, DIN, DPAD, HD2);
  k_prep_wT<<<cdiv(HD  * HD2,  TB), TB, 0, stream>>>(w1b, w1bT, HD2, HD2, HD);
  k_prep_wT<<<cdiv(HD2 * HD,   TB), TB, 0, stream>>>(w2a, w2aT, HD,  HD,  HD2);
  k_prep_wT<<<cdiv(HD  * HD2,  TB), TB, 0, stream>>>(w2b, w2bT, HD2, HD2, HD);
  k_prep_wT<<<cdiv(HD2 * HD,   TB), TB, 0, stream>>>(w3a, w3aT, HD,  HD,  HD2);
  k_prep_wT<<<cdiv(HD  * HD2,  TB), TB, 0, stream>>>(w3b, w3bT, HD2, HD2, HD);
  k_prep_bn<<<1, HD, 0, stream>>>(bn1, scale1, shift1, HD);
  k_prep_bn<<<1, HD, 0, stream>>>(bn2, scale2, shift2, HD);
  k_prep_bn<<<1, HD, 0, stream>>>(bn3, scale3, shift3, HD);

  // degree (with self-loops) -> dinv, shared by all four poolings
  k_deg_init<<<cdiv(NN, TB), TB, 0, stream>>>(dinv, NN);
  k_deg_edge<<<cdiv(EE, TB), TB, 0, stream>>>(dst, dinv, EE);
  k_deg_fin<<<cdiv(NN, TB), TB, 0, stream>>>(dinv, NN);

  const dim3 gW(NN / 16, HD2 / 128);  // (3125, 2) for Ncols=256
  const dim3 gN(NN / 16, HD  / 128);  // (3125, 1) for Ncols=128

  // ---- GIN layer 1 ----
  k_zero_f<<<cdiv(NN * DIN, TB), TB, 0, stream>>>(agg, NN * DIN);
  k_agg_edges<<<cdiv(EE * DIN, TB), TB, 0, stream>>>(xbn, src, dst, agg, EE * DIN, DIN);
  k_add_cvt<<<cdiv(NN * DPAD, TB), TB, 0, stream>>>(xbn, agg, abf, NN * DPAD, DIN, DPAD);
  k_wmma_gemm<0><<<gW, 128, 0, stream>>>(abf, DPAD, w1aT, DPAD, b1a,
                                         nullptr, nullptr, midbf, nullptr, HD2,
                                         NN, DPAD, HD2);
  k_wmma_gemm<1><<<gN, 128, 0, stream>>>(midbf, HD2, w1bT, HD2, b1b,
                                         scale1, shift1, nullptr, h1, HD,
                                         NN, HD2, HD);

  // ---- GIN layer 2 ----
  k_zero_f<<<cdiv(NN * HD, TB), TB, 0, stream>>>(agg, NN * HD);
  k_agg_edges<<<cdiv(EE * HD, TB), TB, 0, stream>>>(h1, src, dst, agg, EE * HD, HD);
  k_add_cvt<<<cdiv(NN * HD, TB), TB, 0, stream>>>(h1, agg, abf, NN * HD, HD, HD);
  k_wmma_gemm<0><<<gW, 128, 0, stream>>>(abf, HD, w2aT, HD, b2a,
                                         nullptr, nullptr, midbf, nullptr, HD2,
                                         NN, HD, HD2);
  k_wmma_gemm<1><<<gN, 128, 0, stream>>>(midbf, HD2, w2bT, HD2, b2b,
                                         scale2, shift2, nullptr, h2, HD,
                                         NN, HD2, HD);

  // ---- GIN layer 3 ----
  k_zero_f<<<cdiv(NN * HD, TB), TB, 0, stream>>>(agg, NN * HD);
  k_agg_edges<<<cdiv(EE * HD, TB), TB, 0, stream>>>(h2, src, dst, agg, EE * HD, HD);
  k_add_cvt<<<cdiv(NN * HD, TB), TB, 0, stream>>>(h2, agg, abf, NN * HD, HD, HD);
  k_wmma_gemm<0><<<gW, 128, 0, stream>>>(abf, HD, w3aT, HD, b3a,
                                         nullptr, nullptr, midbf, nullptr, HD2,
                                         NN, HD, HD2);
  k_wmma_gemm<1><<<gN, 128, 0, stream>>>(midbf, HD2, w3bT, HD2, b3b,
                                         scale3, shift3, nullptr, h3, HD,
                                         NN, HD2, HD);

  // ---- SAG pool 1..3 -> x_local = d_out[0 : G*384) ----
  const float* hs[3]  = {h1, h2, h3};
  const float* wps[3] = {wp1, wp2, wp3};
  const float* bps[3] = {bp1, bp2, bp3};
  for (int p = 0; p < 3; ++p) {
    k_node_dot<<<cdiv(NN, TB), TB, 0, stream>>>(hs[p], wps[p], xw, NN, HD, 0);
    k_score_init<<<cdiv(NN, TB), TB, 0, stream>>>(xw, dinv, sbuf, NN);
    k_score_edge<<<cdiv(EE, TB), TB, 0, stream>>>(xw, dinv, src, dst, sbuf, EE);
    k_zero_f<<<cdiv(GG, TB), TB, 0, stream>>>(zbuf, GG);
    k_score_fin<<<cdiv(NN, TB), TB, 0, stream>>>(sbuf, bps[p], batch, zbuf, NN);
    k_pool_add<<<cdiv(NN * HD, TB), TB, 0, stream>>>(hs[p], sbuf, zbuf, batch,
                                                     out, NN * HD, HD, 3 * HD, p * HD);
  }

  // ---- SAG pool 4 (JumpingKnowledge concat) -> g4 = d_out[G*384 : 2*G*384) ----
  k_node_dot<<<cdiv(NN, TB), TB, 0, stream>>>(h1, wp4,          xw, NN, HD, 0);
  k_node_dot<<<cdiv(NN, TB), TB, 0, stream>>>(h2, wp4 + HD,     xw, NN, HD, 1);
  k_node_dot<<<cdiv(NN, TB), TB, 0, stream>>>(h3, wp4 + 2 * HD, xw, NN, HD, 1);
  k_score_init<<<cdiv(NN, TB), TB, 0, stream>>>(xw, dinv, sbuf, NN);
  k_score_edge<<<cdiv(EE, TB), TB, 0, stream>>>(xw, dinv, src, dst, sbuf, EE);
  k_zero_f<<<cdiv(GG, TB), TB, 0, stream>>>(zbuf, GG);
  k_score_fin<<<cdiv(NN, TB), TB, 0, stream>>>(sbuf, bp4, batch, zbuf, NN);
  float* g4 = out + (size_t)GG * 3 * HD;
  k_pool_add<<<cdiv(NN * HD, TB), TB, 0, stream>>>(h1, sbuf, zbuf, batch,
                                                   g4, NN * HD, HD, 3 * HD, 0);
  k_pool_add<<<cdiv(NN * HD, TB), TB, 0, stream>>>(h2, sbuf, zbuf, batch,
                                                   g4, NN * HD, HD, 3 * HD, HD);
  k_pool_add<<<cdiv(NN * HD, TB), TB, 0, stream>>>(h3, sbuf, zbuf, batch,
                                                   g4, NN * HD, HD, 3 * HD, 2 * HD);
}
